// MeshNetworkARAR_86303072845942
// MI455X (gfx1250) — matrix operand order repaired and checked by reading, not compiled, not run on hardware
//
#include <hip/hip_runtime.h>
#include <hip/hip_bf16.h>

typedef __attribute__((ext_vector_type(16))) __bf16 v16bf;
typedef __attribute__((ext_vector_type(16))) short  v16s;
typedef __attribute__((ext_vector_type(8)))  short  v8s;
typedef __attribute__((ext_vector_type(8)))  float  v8f;
typedef __attribute__((ext_vector_type(4)))  float  f4;

#define NMESH 100000
#define NPATCH 2500000
#define NEDGE 1600000
#define HID 128
#define INDIM 64
#define OUTD 32
#define EPSN 1e-5f

// ---------- helpers ----------
__device__ __forceinline__ unsigned short f32_to_bf16_rne(float f) {
    unsigned int u = __float_as_uint(f);
    u += 0x7FFFu + ((u >> 16) & 1u);
    return (unsigned short)(u >> 16);
}
__device__ __forceinline__ float bf16_bits_to_f32(unsigned short h) {
    return __uint_as_float(((unsigned int)h) << 16);
}

// ---------- zero ----------
__global__ void zero_f32(float* p, long long n) {
    long long i = (long long)blockIdx.x * blockDim.x + threadIdx.x;
    long long stride = (long long)gridDim.x * blockDim.x;
    for (; i < n; i += stride) p[i] = 0.0f;
}

// ---------- segment mean ----------
__global__ void seg_count_kernel(const int* __restrict__ seg, float* __restrict__ cnt) {
    long long g = (long long)blockIdx.x * blockDim.x + threadIdx.x;
    if (g >= NPATCH) return;
    atomicAdd(&cnt[seg[g]], 1.0f);
}
// one thread per (patch, 4-feature chunk)
__global__ void seg_sum_kernel(const float* __restrict__ pf, const int* __restrict__ seg,
                               float* __restrict__ sums) {
    long long g = (long long)blockIdx.x * blockDim.x + threadIdx.x;
    if (g >= (long long)NPATCH * (INDIM / 4)) return;
    int p  = (int)(g >> 4);          // / (64/4)
    int fq = (int)(g & 15);
    f4 v = *(const f4*)&pf[(long long)p * INDIM + 4 * fq];
    float* o = &sums[(long long)seg[p] * INDIM + 4 * fq];
    atomicAdd(o + 0, v.x);
    atomicAdd(o + 1, v.y);
    atomicAdd(o + 2, v.z);
    atomicAdd(o + 3, v.w);
}

// ---------- degrees ----------
__global__ void degree_kernel(const int* __restrict__ src, const int* __restrict__ dst,
                              float* __restrict__ outd, float* __restrict__ ind) {
    long long g = (long long)blockIdx.x * blockDim.x + threadIdx.x;
    if (g >= NEDGE) return;
    atomicAdd(&outd[src[g]], 1.0f);
    atomicAdd(&ind[dst[g]], 1.0f);
}
__global__ void rsqrt_deg_kernel(float* od, float* id) {
    int n = blockIdx.x * blockDim.x + threadIdx.x;
    if (n >= NMESH) return;
    float o = od[n]; o = o < 1.0f ? 1.0f : o;
    float i = id[n]; i = i < 1.0f ? 1.0f : i;
    od[n] = rsqrtf(o);
    id[n] = rsqrtf(i);
}
// feats = (sum/cnt) * inv_sqrt_out   (od already holds rsqrt)
__global__ void feats_final_kernel(float* __restrict__ feats, const float* __restrict__ cnt,
                                   const float* __restrict__ iso) {
    long long g = (long long)blockIdx.x * blockDim.x + threadIdx.x;
    if (g >= (long long)NMESH * INDIM) return;
    int n = (int)(g >> 6);
    float c = cnt[n]; c = c < 1.0f ? 1.0f : c;
    feats[g] = feats[g] / c * iso[n];
}

// ---------- weight prep: f32 -> (bf16 hi, bf16 lo), pre-swizzled into the WMMA
// B-operand lane layout so the GEMM loads one contiguous v16s per lane ----------
// slot index within a matrix: (((kb/32)*8 + wt)*32 + lane)*16 + j
//   half = lane>>4, n = lane&15;  k = kb + half*16 + j;  col = wt*16 + n
__global__ void prep_w_kernel(const float* __restrict__ W1, const float* __restrict__ W2,
                              unsigned short* __restrict__ wbf) {
    const int n1 = INDIM * HID;   // 8192
    const int n2 = HID * HID;     // 16384
    int g = blockIdx.x * blockDim.x + threadIdx.x;
    if (g >= n1 + n2) return;
    const float* Wsrc;
    unsigned short *dhi, *dlo;
    int idx;
    if (g < n1) { Wsrc = W1; dhi = wbf;           dlo = wbf + n1;           idx = g; }
    else        { Wsrc = W2; dhi = wbf + 2 * n1;  dlo = wbf + 2 * n1 + n2;  idx = g - n1; }
    int j    = idx & 15;
    int lane = (idx >> 4) & 31;
    int wt   = (idx >> 9) & 7;
    int kb   = (idx >> 12) * 32;
    int half = lane >> 4;
    int n    = lane & 15;
    int k    = kb + half * 16 + j;
    int col  = wt * 16 + n;
    float v = Wsrc[k * HID + col];
    unsigned short hi = f32_to_bf16_rne(v);
    float lo = v - bf16_bits_to_f32(hi);
    dhi[idx] = hi;
    dlo[idx] = f32_to_bf16_rne(lo);
}

// ---------- edge scatter: agg[dst] += h[src]*ew, float4 granularity ----------
__global__ void scatter_edges_kernel(const int* __restrict__ src, const int* __restrict__ dst,
                                     const float* __restrict__ ew, const float* __restrict__ h,
                                     float* __restrict__ agg, int shift, long long total) {
    long long g = (long long)blockIdx.x * blockDim.x + threadIdx.x;
    if (g >= total) return;
    int qshift = shift - 2;                 // chunks of 4 features
    int qmask  = (1 << qshift) - 1;
    int e  = (int)(g >> qshift);
    int fq = (int)(g & qmask);
    float w = ew[e];
    f4 v = *(const f4*)&h[((long long)src[e] << shift) + 4 * fq];
    float* o = &agg[((long long)dst[e] << shift) + 4 * fq];
    atomicAdd(o + 0, v.x * w);
    atomicAdd(o + 1, v.y * w);
    atomicAdd(o + 2, v.z * w);
    atomicAdd(o + 3, v.w * w);
}

// ---------- WMMA GEMM: out[N x 128] = (A * isi_row) @ W, split-bf16 ----------
// grid = N/16 blocks, 256 threads (8 waves); wave wt -> cols [16wt,16wt+16)
#define LDS_STRIDE 136   // ushorts; 272B rows: 16B-aligned chunks, bank-spread
__global__ __launch_bounds__(256) void gemm_wmma_kernel(
    const float* __restrict__ A, const float* __restrict__ isi,
    const unsigned short* __restrict__ Whi, const unsigned short* __restrict__ Wlo,
    float* __restrict__ out, int K) {
    __shared__ __align__(32) unsigned short lds_hi[16 * LDS_STRIDE];
    __shared__ __align__(32) unsigned short lds_lo[16 * LDS_STRIDE];
    const int baser = blockIdx.x * 16;
    // stage A tile: scale rows by inv_sqrt_in, split into bf16 hi/lo
    for (int i = threadIdx.x; i < 16 * K; i += 256) {
        int r = i / K, c = i - r * K;
        float v = A[(long long)(baser + r) * K + c] * isi[baser + r];
        unsigned short hi = f32_to_bf16_rne(v);
        float lo = v - bf16_bits_to_f32(hi);
        lds_hi[r * LDS_STRIDE + c] = hi;
        lds_lo[r * LDS_STRIDE + c] = f32_to_bf16_rne(lo);
    }
    __syncthreads();

    const int lane = threadIdx.x & 31;
    const int wt   = threadIdx.x >> 5;   // col tile 0..7
    const int m    = lane & 15;
    const int half = lane >> 4;
    const v16s* Bh = reinterpret_cast<const v16s*>(Whi);
    const v16s* Bl = reinterpret_cast<const v16s*>(Wlo);

    v8f acc = {};
    for (int kb = 0; kb < K; kb += 32) {
        // A: two contiguous 16B runs per lane -> ds_load_b128 x2 per operand
        const int base0 = m * LDS_STRIDE + kb + half * 16;
        v8s ah0 = *(const v8s*)&lds_hi[base0];
        v8s ah1 = *(const v8s*)&lds_hi[base0 + 8];
        v8s al0 = *(const v8s*)&lds_lo[base0];
        v8s al1 = *(const v8s*)&lds_lo[base0 + 8];
        v16s ah = __builtin_shufflevector(ah0, ah1, 0,1,2,3,4,5,6,7,8,9,10,11,12,13,14,15);
        v16s al = __builtin_shufflevector(al0, al1, 0,1,2,3,4,5,6,7,8,9,10,11,12,13,14,15);
        // B: one contiguous 32B vector per lane (pre-swizzled)
        const int bidx = (((kb >> 5) * 8) + wt) * 32 + lane;
        v16s bh = Bh[bidx];
        v16s bl = Bl[bidx];

        v16bf a_hi = __builtin_bit_cast(v16bf, ah);
        v16bf a_lo = __builtin_bit_cast(v16bf, al);
        v16bf b_hi = __builtin_bit_cast(v16bf, bh);
        v16bf b_lo = __builtin_bit_cast(v16bf, bl);
        acc = __builtin_amdgcn_wmma_f32_16x16x32_bf16(false, a_hi, false, b_hi, (short)0, acc, false, false);
        acc = __builtin_amdgcn_wmma_f32_16x16x32_bf16(false, a_hi, false, b_lo, (short)0, acc, false, false);
        acc = __builtin_amdgcn_wmma_f32_16x16x32_bf16(false, a_lo, false, b_hi, (short)0, acc, false, false);
    }
    // C/D layout: VGPR v -> row = v + 8*half, col = lane&15 within tile
    const int col = wt * 16 + m;
    #pragma unroll
    for (int v = 0; v < 8; ++v) {
        int row = baser + v + half * 8;
        out[(long long)row * HID + col] = acc[v];
    }
}

// ---------- column reductions (sum / sum-of-squares), per-thread partials ----------
__global__ void col_reduce_kernel(const float* __restrict__ x, float* __restrict__ out,
                                  int nrows, int C, int square) {
    int g = blockIdx.x * blockDim.x + threadIdx.x;
    int T = gridDim.x * blockDim.x;
    int c = g % C;
    int r0 = g / C;
    int step = T / C;
    float acc = 0.0f;
    for (int r = r0; r < nrows; r += step) {
        float v = x[(long long)r * C + c];
        acc += square ? v * v : v;
    }
    atomicAdd(&out[c], acc);
}

// ---------- graph norm passes ----------
__global__ void center_kernel(float* __restrict__ x, const float* __restrict__ s,
                              const float* __restrict__ ms, float invN, long long total) {
    long long g = (long long)blockIdx.x * blockDim.x + threadIdx.x;
    if (g >= total) return;
    int c = (int)(g & (HID - 1));
    x[g] = x[g] - ms[c] * (s[c] * invN);
}
__global__ void normact_kernel(float* __restrict__ x, const float* __restrict__ ss,
                               const float* __restrict__ w, const float* __restrict__ b,
                               float invN, long long total) {
    long long g = (long long)blockIdx.x * blockDim.x + threadIdx.x;
    if (g >= total) return;
    int c = (int)(g & (HID - 1));
    float y = w[c] * x[g] * rsqrtf(ss[c] * invN + EPSN) + b[c];
    x[g] = y > 0.0f ? y : 0.01f * y;
}
__global__ void scale_rows_kernel(float* __restrict__ x, const float* __restrict__ iso,
                                  long long total) {
    long long g = (long long)blockIdx.x * blockDim.x + threadIdx.x;
    if (g >= total) return;
    int n = (int)(g >> 7);   // /HID
    x[g] *= iso[n];
}

// ---------- final readout: out[32] = [r1/N, r2/N] @ Wc ----------
__global__ void final_kernel(const float* __restrict__ r1, const float* __restrict__ r2,
                             const float* __restrict__ Wc, float* __restrict__ out, float invN) {
    int o = threadIdx.x;
    if (o >= OUTD) return;
    float acc = 0.0f;
    for (int c = 0; c < HID; ++c) acc += (r1[c] * invN) * Wc[c * OUTD + o];
    for (int c = 0; c < HID; ++c) acc += (r2[c] * invN) * Wc[(HID + c) * OUTD + o];
    out[o] = acc;
}

extern "C" void kernel_launch(void* const* d_in, const int* in_sizes, int n_in,
                              void* d_out, int out_size, void* d_ws, size_t ws_size,
                              hipStream_t stream) {
    const float* patch_feats = (const float*)d_in[0];
    const int*   seg         = (const int*)d_in[1];
    const int*   src         = (const int*)d_in[2];
    const int*   dst         = (const int*)d_in[3];
    const float* ew          = (const float*)d_in[4];
    const float* W1          = (const float*)d_in[5];
    const float* W2          = (const float*)d_in[6];
    const float* Wc          = (const float*)d_in[7];
    const float* gn1_w       = (const float*)d_in[8];
    const float* gn1_b       = (const float*)d_in[9];
    const float* gn1_ms      = (const float*)d_in[10];
    const float* gn2_w       = (const float*)d_in[11];
    const float* gn2_b       = (const float*)d_in[12];
    const float* gn2_ms      = (const float*)d_in[13];

    const long long N = NMESH;
    float* W = (float*)d_ws;
    // workspace layout (floats)
    const long long o_cnt   = 0;
    const long long o_od    = N;              // out-degree -> inv_sqrt_out
    const long long o_id    = 2 * N;          // in-degree  -> inv_sqrt_in
    const long long o_stats = 3 * N;          // 1024 floats of stats
    const long long o_feats = 3 * N + 1024;   // 64N (also start of h2 reuse: 128N)
    const long long o_agg1  = o_feats + 64 * N;
    const long long o_h1    = o_agg1 + 64 * N;
    const long long o_agg2  = o_h1 + 128 * N;
    const long long o_wbf   = o_agg2 + 128 * N;  // 49152 ushorts, 32B-aligned
    unsigned short* wbf = (unsigned short*)(W + o_wbf);

    float* s1  = W + o_stats + 0;
    float* ss1 = W + o_stats + 128;
    float* r1  = W + o_stats + 256;
    float* s2  = W + o_stats + 384;
    float* ss2 = W + o_stats + 512;
    float* r2  = W + o_stats + 640;

    const float invN = 1.0f / (float)N;
    const long long totE1q = (long long)NEDGE * (INDIM / 4);
    const long long totE2q = (long long)NEDGE * (HID / 4);
    const long long totN1 = (long long)N * INDIM;
    const long long totN2 = (long long)N * HID;

    // 1) zero accumulators (cnt, degs, stats, feats-sums, agg1) and agg2
    zero_f32<<<2048, 256, 0, stream>>>(W, 131 * N + 1024);
    zero_f32<<<2048, 256, 0, stream>>>(W + o_agg2, 128 * N);

    // 2) segment mean + degrees
    seg_count_kernel<<<(NPATCH + 255) / 256, 256, 0, stream>>>(seg, W + o_cnt);
    seg_sum_kernel<<<(unsigned)(((long long)NPATCH * (INDIM / 4) + 255) / 256), 256, 0, stream>>>(
        patch_feats, seg, W + o_feats);
    degree_kernel<<<(NEDGE + 255) / 256, 256, 0, stream>>>(src, dst, W + o_od, W + o_id);
    rsqrt_deg_kernel<<<(NMESH + 255) / 256, 256, 0, stream>>>(W + o_od, W + o_id);
    feats_final_kernel<<<(unsigned)((totN1 + 255) / 256), 256, 0, stream>>>(
        W + o_feats, W + o_cnt, W + o_od);

    // 3) weights -> split bf16, pre-swizzled to B-operand layout
    prep_w_kernel<<<(INDIM * HID + HID * HID + 255) / 256, 256, 0, stream>>>(W1, W2, wbf);

    // 4) layer 1: scatter, WMMA GEMM, GraphNorm + leaky, readout, scale for next conv
    scatter_edges_kernel<<<(unsigned)((totE1q + 255) / 256), 256, 0, stream>>>(
        src, dst, ew, W + o_feats, W + o_agg1, 6, totE1q);
    gemm_wmma_kernel<<<NMESH / 16, 256, 0, stream>>>(
        W + o_agg1, W + o_id, wbf, wbf + 8192, W + o_h1, INDIM);
    col_reduce_kernel<<<128, 256, 0, stream>>>(W + o_h1, s1, NMESH, HID, 0);
    center_kernel<<<(unsigned)((totN2 + 255) / 256), 256, 0, stream>>>(W + o_h1, s1, gn1_ms, invN, totN2);
    col_reduce_kernel<<<128, 256, 0, stream>>>(W + o_h1, ss1, NMESH, HID, 1);
    normact_kernel<<<(unsigned)((totN2 + 255) / 256), 256, 0, stream>>>(W + o_h1, ss1, gn1_w, gn1_b, invN, totN2);
    col_reduce_kernel<<<128, 256, 0, stream>>>(W + o_h1, r1, NMESH, HID, 0);
    scale_rows_kernel<<<(unsigned)((totN2 + 255) / 256), 256, 0, stream>>>(W + o_h1, W + o_od, totN2);

    // 5) layer 2 (h2 reuses feats+agg1 region, which is dead by now)
    scatter_edges_kernel<<<(unsigned)((totE2q + 255) / 256), 256, 0, stream>>>(
        src, dst, ew, W + o_h1, W + o_agg2, 7, totE2q);
    gemm_wmma_kernel<<<NMESH / 16, 256, 0, stream>>>(
        W + o_agg2, W + o_id, wbf + 16384, wbf + 32768, W + o_feats, HID);
    col_reduce_kernel<<<128, 256, 0, stream>>>(W + o_feats, s2, NMESH, HID, 0);
    center_kernel<<<(unsigned)((totN2 + 255) / 256), 256, 0, stream>>>(W + o_feats, s2, gn2_ms, invN, totN2);
    col_reduce_kernel<<<128, 256, 0, stream>>>(W + o_feats, ss2, NMESH, HID, 1);
    normact_kernel<<<(unsigned)((totN2 + 255) / 256), 256, 0, stream>>>(W + o_feats, ss2, gn2_w, gn2_b, invN, totN2);
    col_reduce_kernel<<<128, 256, 0, stream>>>(W + o_feats, r2, NMESH, HID, 0);

    // 6) final readout
    final_kernel<<<1, 32, 0, stream>>>(r1, r2, Wc, (float*)d_out, invN);
}
